// EnhancedContrast_48747878809899
// MI455X (gfx1250) — matrix operand order repaired and checked by reading, not compiled, not run on hardware
//
#include <hip/hip_runtime.h>
#include <hip/hip_bf16.h>

// ---------------------------------------------------------------------------
// EnhancedContrast fused pipeline for MI455X (gfx1250, wave32, WMMA bf16)
//   N=8192, H=512, D=256, G=64, group size = 128 (aligned with 128x128 tiles)
// Symmetric-pair m-pass (halves pos HBM traffic), async-to-LDS staging
// (GLOBAL_LOAD_ASYNC_TO_LDS_B128 + s_wait_asynccnt), pos prefetch.
// ---------------------------------------------------------------------------

typedef __attribute__((ext_vector_type(16))) __bf16 v16bf;
typedef __attribute__((ext_vector_type(8)))  float  v8f;
typedef __attribute__((ext_vector_type(4)))  int    v4i;

union BF16x16 { v16bf v; unsigned short u[16]; };

#define N_   8192
#define H_   512
#define D_   256
#define G_   64
#define TILE 128
#define KSTEP 32
#define INV_TAU 2.0f
#define LAM  0.5f
#define ALPHA_ 1.0f
#define BETA_  1.0f
#define EPS_G 1e-6f
#define EPS_L 1e-5f

#if defined(__has_builtin)
#if __has_builtin(__builtin_amdgcn_global_load_async_to_lds_b128) && \
    __has_builtin(__builtin_amdgcn_s_wait_asynccnt)
#define USE_ASYNC_LDS 1
#endif
#endif

#ifdef USE_ASYNC_LDS
typedef __attribute__((address_space(1))) v4i gv4i;   // global 128-bit chunk
typedef __attribute__((address_space(3))) v4i lv4i;   // LDS    128-bit chunk
#endif

// ------------------------- workspace layout (bytes) ------------------------
#define OFF_X      0u            // bf16 [8192][512]  (input / ping-pong)
#define OFF_H      8388608u      // bf16 [8192][512]  (hidden)
#define OFF_W1T    16777216u     // bf16 [512][512]   (W1 transposed)
#define OFF_W2T    17301504u     // bf16 [512][512]
#define OFF_W3T    17825792u     // bf16 [256][512]   (W3 transposed)
#define OFF_ZAN    18087936u     // bf16 [8192][256]  normalized za_p
#define OFF_ZBN    22282240u     // bf16 [8192][256]  normalized zb_p
#define OFF_ROWSUM 26476544u     // f32 [8192]
#define OFF_ROWPOS 26509312u     // f32 [8192]
#define OFF_COLSUM 26542080u     // f32 [8192]
#define OFF_COLPOS 26574848u     // f32 [8192]
#define OFF_DIAG   26607616u     // f32 [8192]
#define OFF_BSIM   26640384u     // f32 [64][64]
#define STATS_BYTES (5u*32768u + 16384u)

__device__ __forceinline__ unsigned short f2bf(float f) {
    unsigned int u = __builtin_bit_cast(unsigned int, f);
    u += 0x7FFFu + ((u >> 16) & 1u);            // round-to-nearest-even
    return (unsigned short)(u >> 16);
}

// ---------------- 128x32 bf16 strip staging: global -> LDS -----------------
__device__ __forceinline__ void stage_strip(unsigned short (*dst)[KSTEP],
                                            const unsigned short* __restrict__ src,
                                            int rowBase, int K, int k0, int tid) {
#ifdef USE_ASYNC_LDS
#pragma unroll
    for (int v = tid; v < (TILE * KSTEP) / 8; v += 256) {
        int row = v >> 2, q = v & 3;
        __builtin_amdgcn_global_load_async_to_lds_b128(
            (gv4i*)(src + (size_t)(rowBase + row) * K + k0 + q * 8),
            (lv4i*)&dst[row][q * 8], 0, 0);
    }
#else
#pragma unroll
    for (int v = tid; v < (TILE * KSTEP) / 8; v += 256) {
        int row = v >> 2, q = v & 3;
        *(uint4*)&dst[row][q * 8] =
            *(const uint4*)(src + (size_t)(rowBase + row) * K + k0 + q * 8);
    }
#endif
}

__device__ __forceinline__ void stage_wait() {
#ifdef USE_ASYNC_LDS
    __builtin_amdgcn_s_wait_asynccnt(0);
#endif
}

// ------- WMMA fragment loads (ISA 7.12.2 layouts, 16x16x32 bf16) -----------
__device__ __forceinline__ v16bf load_afrag(const unsigned short (*lds)[KSTEP],
                                            int wave, int lane) {
    BF16x16 af;
    const int arow = wave * 16 + (lane & 15);
    const int kaoff = (lane & 16) ? 8 : 0;
#pragma unroll
    for (int p = 0; p < 8; ++p) {
        int kp = ((p < 4) ? 2 * p : 16 + 2 * (p - 4)) + kaoff;
        af.u[2 * p]     = lds[arow][kp];
        af.u[2 * p + 1] = lds[arow][kp + 1];
    }
    return af.v;
}

__device__ __forceinline__ v16bf load_bfrag(const unsigned short (*lds)[KSTEP],
                                            int tc, int lane) {
    BF16x16 bfv;
    const int bcol = tc * 16 + (lane & 15);
    const int kboff = (lane & 16) ? 8 : 0;
#pragma unroll
    for (int p = 0; p < 8; ++p) {
        int k = ((p < 4) ? 2 * p : 16 + 2 * (p - 4)) + kboff;
        bfv.u[2 * p]     = lds[bcol][k];
        bfv.u[2 * p + 1] = lds[bcol][k + 1];
    }
    return bfv.v;
}

// ------------------------- fp32 -> bf16 convert ----------------------------
__global__ void k_convert(const float* __restrict__ src,
                          unsigned short* __restrict__ dst, int n) {
    int i = blockIdx.x * blockDim.x + threadIdx.x;
    if (i < n) dst[i] = f2bf(src[i]);
}

// ------------------ fp32 [K][Nc] -> bf16 transposed [Nc][K] ----------------
__global__ void k_transpose_bf16(const float* __restrict__ src,
                                 unsigned short* __restrict__ dst,
                                 int K, int Nc) {
    int i = blockIdx.x * blockDim.x + threadIdx.x;
    if (i < K * Nc) {
        int n = i / K, k = i % K;
        dst[i] = f2bf(src[k * Nc + n]);
    }
}

// ---------------------------------------------------------------------------
// Tiled bf16 WMMA GEMM: C[M][Nc] = A[M][K] * Bt[Nc][K]^T  (Bt row-major in K)
// mode 0: DyT epilogue  -> bf16 out        (h = tanh(a*(x+b))*g + be)
// mode 1: SiLU epilogue -> f32 to d_out    (p = x+b; p*sigmoid(p))
// ---------------------------------------------------------------------------
__global__ __launch_bounds__(256)
void k_gemm_bf16(const unsigned short* __restrict__ A,
                 const unsigned short* __restrict__ Bt,
                 int K, int Nc, int mode,
                 const float* __restrict__ bias,
                 const float* __restrict__ alpha,
                 const float* __restrict__ gamma,
                 const float* __restrict__ beta,
                 unsigned short* __restrict__ outBf,
                 float* __restrict__ outF, int outStride, int outColOff) {
    __shared__ __align__(16) unsigned short lA[TILE][KSTEP];
    __shared__ __align__(16) unsigned short lB[TILE][KSTEP];

    const int bi = blockIdx.x, bj = blockIdx.y;
    const int tid = threadIdx.x;
    const int wave = tid >> 5, lane = tid & 31;

    v8f acc[8];
#pragma unroll
    for (int t = 0; t < 8; ++t)
#pragma unroll
        for (int e = 0; e < 8; ++e) acc[t][e] = 0.0f;

    for (int k0 = 0; k0 < K; k0 += KSTEP) {
        stage_strip(lA, A, bi * TILE, K, k0, tid);
        stage_strip(lB, Bt, bj * TILE, K, k0, tid);
        stage_wait();
        __syncthreads();

        const v16bf av = load_afrag(lA, wave, lane);
#pragma unroll
        for (int tc = 0; tc < 8; ++tc) {
            acc[tc] = __builtin_amdgcn_wmma_f32_16x16x32_bf16(
                false, av, false, load_bfrag(lB, tc, lane), (short)0,
                acc[tc], false, false);
        }
        __syncthreads();
    }

    const float al = (mode == 0) ? alpha[0] : 0.0f;
#pragma unroll
    for (int tc = 0; tc < 8; ++tc) {
        const int j = bj * TILE + tc * 16 + (lane & 15);
#pragma unroll
        for (int r = 0; r < 8; ++r) {
            const int i = bi * TILE + wave * 16 + r + ((lane & 16) ? 8 : 0);
            float x = acc[tc][r] + bias[j];
            if (mode == 0) {
                float h = tanhf(al * x) * gamma[j] + beta[j];
                outBf[(size_t)i * Nc + j] = f2bf(h);
            } else {
                float s = x / (1.0f + __expf(-x));   // x * sigmoid(x)
                outF[(size_t)i * outStride + outColOff + j] = s;
            }
        }
    }
}

// --------- per-row L2 norm + normalize to bf16 (one wave per row) ----------
__global__ __launch_bounds__(256)
void k_rownorm(const float* __restrict__ srcBase, int colOff,
               unsigned short* __restrict__ dst) {
    const int wave = threadIdx.x >> 5, lane = threadIdx.x & 31;
    const int row = blockIdx.x * 8 + wave;
    const float* src = srcBase + (size_t)row * (2 * D_) + colOff;
    float s = 0.0f;
    float v[8];
#pragma unroll
    for (int e = 0; e < 8; ++e) { v[e] = src[lane + e * 32]; s += v[e] * v[e]; }
#pragma unroll
    for (int m = 16; m > 0; m >>= 1) s += __shfl_xor(s, m, 32);
    const float inv = 1.0f / sqrtf(s);
#pragma unroll
    for (int e = 0; e < 8; ++e)
        dst[(size_t)row * D_ + lane + e * 32] = f2bf(v[e] * inv);
}

// ---------------- per-tile fused reductions of the m matrix ----------------
__device__ __forceinline__ void reduce_tile(
    const v8f* acc, int rb, int cb, int wave, int lane, int tid,
    const float* __restrict__ pos,
    float* ls_rs, float* ls_rp, float* ls_cs, float* ls_cp, float* ls_tile,
    float* __restrict__ rowsum, float* __restrict__ rowpos,
    float* __restrict__ colsum, float* __restrict__ colpos,
    float* __restrict__ diag,   float* __restrict__ bsim) {
    if (tid < TILE) { ls_rs[tid] = 0.f; ls_rp[tid] = 0.f; ls_cs[tid] = 0.f; ls_cp[tid] = 0.f; }
    if (tid == 0) *ls_tile = 0.f;
    __syncthreads();

    float tsum = 0.0f;
#pragma unroll
    for (int tc = 0; tc < 8; ++tc) {
        const int jl = tc * 16 + (lane & 15);
        const int gj = cb * TILE + jl;
#pragma unroll
        for (int r = 0; r < 8; ++r) {
            const int il = wave * 16 + r + ((lane & 16) ? 8 : 0);
            const int gi = rb * TILE + il;
            const float e = __expf(acc[tc][r] * INV_TAU);
            const float pij = pos[(size_t)gi * N_ + gj];
            const float pji = pos[(size_t)gj * N_ + gi];
            atomicAdd(&ls_rs[il], e);
            atomicAdd(&ls_rp[il], e * pij);
            atomicAdd(&ls_cs[jl], e);
            atomicAdd(&ls_cp[jl], e * pji);
            tsum += e;
            if (gi == gj) diag[gi] = e;     // only possible on rb==cb tiles
        }
    }
    atomicAdd(ls_tile, tsum);
    __syncthreads();

    if (tid < TILE) {
        atomicAdd(&rowsum[rb * TILE + tid], ls_rs[tid]);
        atomicAdd(&rowpos[rb * TILE + tid], ls_rp[tid]);
        atomicAdd(&colsum[cb * TILE + tid], ls_cs[tid]);
        atomicAdd(&colpos[cb * TILE + tid], ls_cp[tid]);
    }
    if (tid == 0) atomicAdd(&bsim[rb * G_ + cb], *ls_tile);
    __syncthreads();
}

// ---------------------------------------------------------------------------
// Symmetric-pair fused m-pass: one block computes tiles (bi,bj) AND (bj,bi),
// so both uses of each 64KB pos sub-block (as pos[i,j] and as pos[j,i]) stay
// local to one block: pos HBM traffic drops 2x (512MB -> ~266MB), which is
// the dominant term of the whole pipeline at 23.3 TB/s.
// m[i,j] = exp(2*dot(zan_i, zbn_j)) is computed by WMMA and never stored.
// Tile size 128 == group size, so tile (rb,cb) -> one bsim[rb][cb] atomic.
// ---------------------------------------------------------------------------
__global__ __launch_bounds__(256)
void k_mtile_pair(const unsigned short* __restrict__ zan,
                  const unsigned short* __restrict__ zbn,
                  const float* __restrict__ pos,
                  float* __restrict__ rowsum, float* __restrict__ rowpos,
                  float* __restrict__ colsum, float* __restrict__ colpos,
                  float* __restrict__ diag,   float* __restrict__ bsim) {
    __shared__ __align__(16) unsigned short sAI[TILE][KSTEP];  // zan rows I
    __shared__ __align__(16) unsigned short sBJ[TILE][KSTEP];  // zbn rows J
    __shared__ __align__(16) unsigned short sAJ[TILE][KSTEP];  // zan rows J
    __shared__ __align__(16) unsigned short sBI[TILE][KSTEP];  // zbn rows I
    __shared__ float ls_rs[TILE], ls_rp[TILE], ls_cs[TILE], ls_cp[TILE];
    __shared__ float ls_tile;

    // unrank upper-triangular pair index -> (bi <= bj)
    int p = blockIdx.x;
    int bi = 0;
    while (p >= G_ - bi) { p -= G_ - bi; ++bi; }
    const int bj = bi + p;
    const bool diagBlk = (bi == bj);

    const int tid = threadIdx.x;
    const int wave = tid >> 5, lane = tid & 31;

    // warm L2/WGP$ with both pos sub-blocks while the WMMA phase runs
#pragma unroll
    for (int v = tid; v < 512; v += 256) {
        int row = v >> 2, seg = v & 3;
        __builtin_prefetch(pos + (size_t)(bi * TILE + row) * N_ + bj * TILE + seg * 32, 0, 0);
        __builtin_prefetch(pos + (size_t)(bj * TILE + row) * N_ + bi * TILE + seg * 32, 0, 0);
    }

    v8f accP[8], accQ[8];
#pragma unroll
    for (int t = 0; t < 8; ++t)
#pragma unroll
        for (int e = 0; e < 8; ++e) { accP[t][e] = 0.0f; accQ[t][e] = 0.0f; }

    for (int k0 = 0; k0 < D_; k0 += KSTEP) {
        stage_strip(sAI, zan, bi * TILE, D_, k0, tid);
        stage_strip(sBJ, zbn, bj * TILE, D_, k0, tid);
        if (!diagBlk) {
            stage_strip(sAJ, zan, bj * TILE, D_, k0, tid);
            stage_strip(sBI, zbn, bi * TILE, D_, k0, tid);
        }
        stage_wait();
        __syncthreads();

        const v16bf aI = load_afrag(sAI, wave, lane);
#pragma unroll
        for (int tc = 0; tc < 8; ++tc) {
            accP[tc] = __builtin_amdgcn_wmma_f32_16x16x32_bf16(
                false, aI, false, load_bfrag(sBJ, tc, lane), (short)0,
                accP[tc], false, false);
        }
        if (!diagBlk) {
            const v16bf aJ = load_afrag(sAJ, wave, lane);
#pragma unroll
            for (int tc = 0; tc < 8; ++tc) {
                accQ[tc] = __builtin_amdgcn_wmma_f32_16x16x32_bf16(
                    false, aJ, false, load_bfrag(sBI, tc, lane), (short)0,
                    accQ[tc], false, false);
            }
        }
        __syncthreads();
    }

    reduce_tile(accP, bi, bj, wave, lane, tid, pos,
                ls_rs, ls_rp, ls_cs, ls_cp, &ls_tile,
                rowsum, rowpos, colsum, colpos, diag, bsim);
    if (!diagBlk)
        reduce_tile(accQ, bj, bi, wave, lane, tid, pos,
                    ls_rs, ls_rp, ls_cs, ls_cp, &ls_tile,
                    rowsum, rowpos, colsum, colpos, diag, bsim);
}

// --------------------------- loss finalization -----------------------------
__global__ __launch_bounds__(256)
void k_final(const float* __restrict__ rowsum, const float* __restrict__ rowpos,
             const float* __restrict__ colsum, const float* __restrict__ colpos,
             const float* __restrict__ diag,   const float* __restrict__ bsim,
             float* __restrict__ out) {
    __shared__ float accA, accB, acc0, acc1, accIn;
    __shared__ float psg[G_];
    const int tid = threadIdx.x;
    if (tid == 0) { accA = 0.f; accB = 0.f; acc0 = 0.f; acc1 = 0.f; accIn = 0.f; }
    __syncthreads();

    float sA = 0.f, sB = 0.f;
    for (int i = tid; i < N_; i += 256) {
        sA += logf(rowpos[i] / (rowsum[i] + EPS_G));
        sB += logf(colpos[i] / (colsum[i] + EPS_G));
    }
    atomicAdd(&accA, sA);
    atomicAdd(&accB, sB);

    if (tid < G_) {
        float s = 0.f;
        for (int r = 0; r < 128; ++r) s += diag[tid * 128 + r];
        psg[tid] = s;
    }
    __syncthreads();

    if (tid < G_) {
        const float gs = bsim[tid * G_ + tid];
        float n0 = 0.f, n1 = 0.f;
        for (int k = 0; k < G_; ++k) { n0 += bsim[k * G_ + tid]; n1 += bsim[tid * G_ + k]; }
        n0 -= gs; n1 -= gs;
        const float p = psg[tid];
        atomicAdd(&acc0, logf(p / (n0 + EPS_L)));
        atomicAdd(&acc1, logf(p / (n1 + EPS_L)));
        atomicAdd(&accIn, logf(p / (gs - p + EPS_L)));
    }
    __syncthreads();

    if (tid == 0) {
        const float lori_a = -accA / (float)N_;
        const float lori_b = -accB / (float)N_;
        const float global_loss = LAM * lori_a + (1.0f - LAM) * lori_b;
        const float inter = 0.5f * (acc0 / (float)G_ + acc1 / (float)G_);
        const float inner = -accIn / (float)G_;
        out[0] = global_loss + ALPHA_ * inter + BETA_ * inner;
    }
}

// ---------------------------------------------------------------------------
extern "C" void kernel_launch(void* const* d_in, const int* in_sizes, int n_in,
                              void* d_out, int out_size, void* d_ws, size_t ws_size,
                              hipStream_t stream) {
    const float* za  = (const float*)d_in[0];
    const float* zb  = (const float*)d_in[1];
    const float* pos = (const float*)d_in[2];
    // d_in[3] = batch (int32) — groups are i/128 by construction; unused.
    const float* W1  = (const float*)d_in[4];
    const float* b1  = (const float*)d_in[5];
    const float* a1  = (const float*)d_in[6];
    const float* g1  = (const float*)d_in[7];
    const float* be1 = (const float*)d_in[8];
    const float* W2  = (const float*)d_in[9];
    const float* b2  = (const float*)d_in[10];
    const float* a2  = (const float*)d_in[11];
    const float* g2  = (const float*)d_in[12];
    const float* be2 = (const float*)d_in[13];
    const float* W3  = (const float*)d_in[14];
    const float* b3  = (const float*)d_in[15];

    float* out  = (float*)d_out;
    float* outP = out + 1;                       // [8192][512] = za_p | zb_p
    char* ws = (char*)d_ws;

    unsigned short* X    = (unsigned short*)(ws + OFF_X);
    unsigned short* Hbuf = (unsigned short*)(ws + OFF_H);
    unsigned short* W1t  = (unsigned short*)(ws + OFF_W1T);
    unsigned short* W2t  = (unsigned short*)(ws + OFF_W2T);
    unsigned short* W3t  = (unsigned short*)(ws + OFF_W3T);
    unsigned short* zan  = (unsigned short*)(ws + OFF_ZAN);
    unsigned short* zbn  = (unsigned short*)(ws + OFF_ZBN);
    float* rowsum = (float*)(ws + OFF_ROWSUM);
    float* rowpos = (float*)(ws + OFF_ROWPOS);
    float* colsum = (float*)(ws + OFF_COLSUM);
    float* colpos = (float*)(ws + OFF_COLPOS);
    float* diagv  = (float*)(ws + OFF_DIAG);
    float* bsim   = (float*)(ws + OFF_BSIM);

    (void)hipMemsetAsync(ws + OFF_ROWSUM, 0, STATS_BYTES, stream);

    // weights -> transposed bf16 (B-tiles then stage identically to A-tiles)
    k_transpose_bf16<<<(H_ * H_ + 255) / 256, 256, 0, stream>>>(W1, W1t, H_, H_);
    k_transpose_bf16<<<(H_ * H_ + 255) / 256, 256, 0, stream>>>(W2, W2t, H_, H_);
    k_transpose_bf16<<<(H_ * D_ + 255) / 256, 256, 0, stream>>>(W3, W3t, H_, D_);

    for (int s = 0; s < 2; ++s) {
        const float* z = s ? zb : za;
        unsigned short* zn = s ? zbn : zan;
        k_convert<<<(N_ * H_ + 255) / 256, 256, 0, stream>>>(z, X, N_ * H_);
        // h1 = DyT(z@W1 + b1)
        k_gemm_bf16<<<dim3(N_ / TILE, H_ / TILE), 256, 0, stream>>>(
            X, W1t, H_, H_, 0, b1, a1, g1, be1, Hbuf, nullptr, 0, 0);
        // h2 = DyT(h1@W2 + b2)   (reuse X)
        k_gemm_bf16<<<dim3(N_ / TILE, H_ / TILE), 256, 0, stream>>>(
            Hbuf, W2t, H_, H_, 0, b2, a2, g2, be2, X, nullptr, 0, 0);
        // p = SiLU(h2@W3 + b3) -> f32 into d_out (interleaved za_p|zb_p)
        k_gemm_bf16<<<dim3(N_ / TILE, D_ / TILE), 256, 0, stream>>>(
            X, W3t, H_, D_, 1, b3, nullptr, nullptr, nullptr,
            nullptr, outP, 2 * D_, s * D_);
        // normalized bf16 copy for the cosine GEMM
        k_rownorm<<<N_ / 8, 256, 0, stream>>>(outP, s * D_, zn);
    }

    // fused symmetric-pair m-pass: 2080 blocks cover all 64x64 tiles
    k_mtile_pair<<<G_ * (G_ + 1) / 2, 256, 0, stream>>>(
        zan, zbn, pos, rowsum, rowpos, colsum, colpos, diagv, bsim);

    k_final<<<1, 256, 0, stream>>>(rowsum, rowpos, colsum, colpos, diagv, bsim, out);
}
